// DeformConvBlock_8280696947121
// MI455X (gfx1250) — matrix operand order repaired and checked by reading, not compiled, not run on hardware
//
#include <hip/hip_runtime.h>

// ---------------------------------------------------------------------------
// Deformable conv2d + bias + ReLU for MI455X (gfx1250).
// Pass 1: x NCHW f32 -> NHWC bf16 (d_ws), weight -> [tap][cout][136] bf16.
// Pass 2: fused bilinear-sample (packed f32 math -> v_pk_fma_f32) + implicit
//         GEMM on v_wmma_f32_16x16x32_bf16; weight tile staged by the Tensor
//         Data Mover (tensor_load_to_lds / TENSORcnt).
// ---------------------------------------------------------------------------

typedef __attribute__((ext_vector_type(16))) __bf16          v16bf;
typedef __attribute__((ext_vector_type(8)))  float           v8f;
typedef __attribute__((ext_vector_type(2)))  float           v2f;
typedef __attribute__((ext_vector_type(8)))  unsigned short  v8us;
typedef __attribute__((ext_vector_type(16))) unsigned short  v16us;
typedef __attribute__((ext_vector_type(4)))  unsigned int    v4u;
typedef __attribute__((ext_vector_type(8)))  int             v8i;
typedef __attribute__((ext_vector_type(4)))  int             v4i;

#define CIN   128
#define COUT  128
#define HH    64
#define WW    64
#define KK    9
#define PLANE (HH * WW)
#define ROWP  136            // padded LDS/ws row stride (128 + 8)

#if __has_builtin(__builtin_amdgcn_tensor_load_to_lds) && \
    __has_builtin(__builtin_amdgcn_s_wait_tensorcnt)
#define USE_TDM 1
#endif

__device__ __forceinline__ unsigned f32_to_bf16_u(float f) {
  unsigned u = __builtin_bit_cast(unsigned, f);
  u += 0x7FFFu + ((u >> 16) & 1u);          // round-to-nearest-even
  return u >> 16;
}
__device__ __forceinline__ unsigned short f32_to_bf16(float f) {
  return (unsigned short)f32_to_bf16_u(f);
}
// one dword = two packed bf16 -> float2 {lo, hi}
__device__ __forceinline__ v2f bf16x2_to_f32x2(unsigned u) {
  v2f r;
  r[0] = __builtin_bit_cast(float, u << 16);
  r[1] = __builtin_bit_cast(float, u & 0xFFFF0000u);
  return r;
}

// ---------------------------------------------------------------------------
// Pass 1: layout/precision conversion.
//   blocks 0..511  : x[b][cin][h][w] f32 -> xt[b][h][w][cin] bf16
//   blocks 512..520: wgt[cout][cin][k]  -> wt[k][cout][ROWP] bf16 (padded)
// ---------------------------------------------------------------------------
__global__ __launch_bounds__(256)
void prep_bf16(const float* __restrict__ x, const float* __restrict__ wgt,
               unsigned short* __restrict__ xt, unsigned short* __restrict__ wt)
{
  const int blk = blockIdx.x;
  if (blk < 512) {
    const int b = blk >> 6, h = blk & 63;
    for (int i = threadIdx.x; i < WW * CIN; i += 256) {
      const int cin = i & 127, w = i >> 7;
      const float v = x[((b * CIN + cin) * HH + h) * WW + w];
      xt[((b * HH + h) * WW + w) * CIN + cin] = f32_to_bf16(v);  // coalesced st
    }
  } else {
    const int k = blk - 512;
    if (k < KK) {
      for (int i = threadIdx.x; i < COUT * CIN; i += 256) {
        const int cin = i & 127, cout = i >> 7;
        wt[(k * COUT + cout) * ROWP + cin] =
            f32_to_bf16(wgt[(cout * CIN + cin) * KK + k]);       // coalesced st
      }
    }
  }
}

// ---------------------------------------------------------------------------
// Pass 2: fused sampling + GEMM.
// ---------------------------------------------------------------------------
__global__ __launch_bounds__(256, 2)
void deform_conv2d_relu_wmma(const unsigned short* __restrict__ xt,
                             const unsigned short* __restrict__ wt,
                             const float* __restrict__ off,
                             const float* __restrict__ bias,
                             float* __restrict__ out)
{
  __shared__ __attribute__((aligned(32))) unsigned short Wl[COUT * ROWP]; // 34 KB
  __shared__ __attribute__((aligned(32))) unsigned short Sl[64   * ROWP]; // 17 KB

  const int tid  = threadIdx.x;
  const int lane = tid & 31;
  const int wave = tid >> 5;       // 8 wave32
  const int wg   = blockIdx.x;     // 512 workgroups
  const int b    = wg >> 6;
  const int h    = wg & 63;

  v8f acc[4] = {};                 // wave: 16 couts x 4 pos-tiles of 16

  const int l16   = lane & 15;
  const int lhalf = lane >> 4;

  for (int kpos = 0; kpos < KK; ++kpos) {
    __syncthreads();   // previous tap's LDS reads complete

    // ---- stage weight tile Wl[cout][cin] via TDM (or plain block copy) ----
#ifdef USE_TDM
    if (wave == 0) {   // wave-uniform branch: single DMA issue
      const unsigned long long ga =
          (unsigned long long)(const void*)(wt + (size_t)kpos * COUT * ROWP);
      const unsigned lds_off = (unsigned)(unsigned long long)(const void*)&Wl[0];
      v4u g0;
      g0[0] = 1u;                                        // count=1 user D#
      g0[1] = lds_off;                                   // lds_addr
      g0[2] = (unsigned)(ga & 0xFFFFFFFFu);              // global_addr lo
      g0[3] = (unsigned)((ga >> 32) & 0x01FFFFFFu) | 0x80000000u; // hi | type=2
      v8i g1 = {};
      g1[0] = 0x00010000;                                // data_size=2B, mask=0
      g1[1] = (int)((ROWP & 0xFFFF) << 16);              // tensor_dim0 lo16
      g1[2] = (int)((COUT & 0xFFFF) << 16);              // td0 hi | tensor_dim1 lo
      g1[3] = (int)(ROWP << 16);                         // td1 hi | tile_dim0
      g1[4] = (int)(COUT & 0xFFFF);                      // tile_dim1 (tile_dim2=0)
      g1[5] = (int)ROWP;                                 // tensor_dim0_stride lo32
      g1[6] = (int)(((ROWP * COUT) & 0xFFFF) << 16);     // td1_stride lo16
      g1[7] = (int)((ROWP * COUT) >> 16);                // td1_stride hi
      v4i z = {};
#if __clang_major__ >= 23
      v8i z8 = {};
      __builtin_amdgcn_tensor_load_to_lds(g0, g1, z, z, z8, 0);
#else
      __builtin_amdgcn_tensor_load_to_lds(g0, g1, z, z, 0);
#endif
    }
#else
    {
      const v8us* src = (const v8us*)(wt + (size_t)kpos * COUT * ROWP);
      v8us* dst = (v8us*)Wl;
      for (int i = tid; i < (COUT * ROWP) / 8; i += 256) dst[i] = src[i];
    }
#endif

    // ---- stage bilinear-sampled tile Sl[pos][cin] from NHWC bf16 x ----
    {
      const int pos  = tid & 63;           // w coordinate
      const int cin0 = (tid >> 6) * 32;    // 4 blocks of 32 channels
      const int ki = kpos / 3, kj = kpos % 3;
      const float oy = off[((b * (2 * KK) + 2 * kpos    ) * HH + h) * WW + pos];
      const float ox = off[((b * (2 * KK) + 2 * kpos + 1) * HH + h) * WW + pos];
      if (kpos + 1 < KK)
        __builtin_prefetch(&off[((b * (2 * KK) + 2 * (kpos + 1)) * HH + h) * WW + pos], 0, 1);
      const float ys = (float)h   + (float)(ki - 1) + oy;
      const float xs = (float)pos + (float)(kj - 1) + ox;
      const float y0f = floorf(ys), x0f = floorf(xs);
      const float wy = ys - y0f,   wx = xs - x0f;
      const int y0 = (int)y0f, x0 = (int)x0f;
      const int y1 = y0 + 1,   x1 = x0 + 1;
      const float vy0 = (y0 >= 0 && y0 < HH) ? 1.f : 0.f;
      const float vy1 = (y1 >= 0 && y1 < HH) ? 1.f : 0.f;
      const float vx0 = (x0 >= 0 && x0 < WW) ? 1.f : 0.f;
      const float vx1 = (x1 >= 0 && x1 < WW) ? 1.f : 0.f;
      const float w00 = (1.f - wy) * (1.f - wx) * vy0 * vx0;
      const float w01 = (1.f - wy) * wx         * vy0 * vx1;
      const float w10 = wy * (1.f - wx)         * vy1 * vx0;
      const float w11 = wy * wx                 * vy1 * vx1;
      const v2f W00 = {w00, w00}, W01 = {w01, w01};
      const v2f W10 = {w10, w10}, W11 = {w11, w11};
      const int y0c = min(max(y0, 0), HH - 1), y1c = min(max(y1, 0), HH - 1);
      const int x0c = min(max(x0, 0), WW - 1), x1c = min(max(x1, 0), WW - 1);
      const unsigned short* xb = xt + (size_t)b * (PLANE * CIN);
      const v4u* p00 = (const v4u*)(xb + (y0c * WW + x0c) * CIN + cin0);
      const v4u* p01 = (const v4u*)(xb + (y0c * WW + x1c) * CIN + cin0);
      const v4u* p10 = (const v4u*)(xb + (y1c * WW + x0c) * CIN + cin0);
      const v4u* p11 = (const v4u*)(xb + (y1c * WW + x1c) * CIN + cin0);
      #pragma unroll
      for (int cc = 0; cc < 4; ++cc) {     // chunks of 8 channels, 16B loads
        const v4u u00 = p00[cc], u01 = p01[cc], u10 = p10[cc], u11 = p11[cc];
        v4u r;
        #pragma unroll
        for (int j = 0; j < 4; ++j) {      // dword = 2 channels, packed f32 math
          v2f v = W00 * bf16x2_to_f32x2(u00[j]);
          v = __builtin_elementwise_fma(W01, bf16x2_to_f32x2(u01[j]), v);
          v = __builtin_elementwise_fma(W10, bf16x2_to_f32x2(u10[j]), v);
          v = __builtin_elementwise_fma(W11, bf16x2_to_f32x2(u11[j]), v);
          r[j] = (f32_to_bf16_u(v[1]) << 16) | f32_to_bf16_u(v[0]);
        }
        *(v4u*)&Sl[pos * ROWP + cin0 + cc * 8] = r;
      }
    }

#ifdef USE_TDM
    __builtin_amdgcn_s_wait_tensorcnt(0);  // weight DMA complete (wave 0)
#endif
    __syncthreads();                        // tiles visible to all waves

    // ---- GEMM: 4 cin chunks of 32; A=weights(16x32), B=sampled(32x16) ----
    const int coutBase = wave * 16;
    #pragma unroll
    for (int c = 0; c < 4; ++c) {
      const int cinc = c * 32;
      // A fragment: lanes 0-15 K 0..7/16..23; lanes 16-31 K 8..15/24..31
      const unsigned short* arow =
          &Wl[(coutBase + l16) * ROWP + cinc + lhalf * 8];
      const v8us alo = *(const v8us*)(arow);
      const v8us ahi = *(const v8us*)(arow + 16);
      v16us av;
      #pragma unroll
      for (int i = 0; i < 8; ++i) { av[i] = alo[i]; av[i + 8] = ahi[i]; }
      const v16bf a = __builtin_bit_cast(v16bf, av);

      // Preload all 4 B fragments, then issue WMMAs back-to-back.
      v16bf bf[4];
      #pragma unroll
      for (int t = 0; t < 4; ++t) {
        const unsigned short* brow =
            &Sl[(t * 16 + l16) * ROWP + cinc + lhalf * 8];
        const v8us blo = *(const v8us*)(brow);
        const v8us bhi = *(const v8us*)(brow + 16);
        v16us bv;
        #pragma unroll
        for (int i = 0; i < 8; ++i) { bv[i] = blo[i]; bv[i + 8] = bhi[i]; }
        bf[t] = __builtin_bit_cast(v16bf, bv);
      }
      #pragma unroll
      for (int t = 0; t < 4; ++t)
        acc[t] = __builtin_amdgcn_wmma_f32_16x16x32_bf16(
            false, a, false, bf[t], (short)0, acc[t], false, false);
    }
  }

  // ---- epilogue: bias + ReLU + coalesced f32 stores ----
  const int coutBase = wave * 16;
  const int rowAdd   = lhalf * 8;
  float* ob = out + b * (COUT * PLANE) + h * WW;
  #pragma unroll
  for (int t = 0; t < 4; ++t) {
    const int wcol = t * 16 + l16;
    #pragma unroll
    for (int r = 0; r < 8; ++r) {
      const int cout = coutBase + rowAdd + r;
      float v = acc[t][r] + bias[cout];
      v = v > 0.f ? v : 0.f;
      ob[cout * PLANE + wcol] = v;
    }
  }
}

extern "C" void kernel_launch(void* const* d_in, const int* in_sizes, int n_in,
                              void* d_out, int out_size, void* d_ws, size_t ws_size,
                              hipStream_t stream) {
  (void)in_sizes; (void)n_in; (void)ws_size; (void)out_size;
  const float* x    = (const float*)d_in[0];
  const float* off  = (const float*)d_in[1];
  const float* wgt  = (const float*)d_in[2];
  const float* bias = (const float*)d_in[3];
  float* out        = (float*)d_out;

  unsigned short* xt = (unsigned short*)d_ws;                 // 8 MB NHWC bf16
  unsigned short* wt = xt + (size_t)8 * PLANE * CIN;          // 306 KB weights

  prep_bf16<<<dim3(512 + KK), dim3(256), 0, stream>>>(x, wgt, xt, wt);
  deform_conv2d_relu_wmma<<<dim3(512), dim3(256), 0, stream>>>(xt, wt, off, bias, out);
}
